// SelfAttention_52785148068645
// MI455X (gfx1250) — compile-verified, matrix operand order checked by reading, not run on hardware
//
#include <hip/hip_runtime.h>
#include <hip/hip_bf16.h>
#include <stdint.h>

#define BB 4
#define SS 2048
#define HH 1024
#define NHH 16
#define HSS 64
#define MM (BB * SS)   // 8192 rows

typedef __attribute__((ext_vector_type(16))) __bf16 v16bf;
typedef __attribute__((ext_vector_type(8)))  float  v8f;
typedef __attribute__((ext_vector_type(4)))  uint32_t v4u;
typedef __attribute__((ext_vector_type(4)))  int      v4i;
typedef __attribute__((ext_vector_type(8)))  int      v8i;

#if defined(__gfx1250__) && __has_builtin(__builtin_amdgcn_tensor_load_to_lds) && __has_builtin(__builtin_amdgcn_s_wait_tensorcnt)
#define HAVE_TDM 1
#else
#define HAVE_TDM 0
#endif

// ---------- helpers ----------

__device__ __forceinline__ uint16_t f2bf(float f) {
    uint32_t u = __builtin_bit_cast(uint32_t, f);
    uint32_t r = u + 0x7FFFu + ((u >> 16) & 1u);   // round-to-nearest-even
    return (uint16_t)(r >> 16);
}

struct U128x2 { uint4 a, b; };

__device__ __forceinline__ v16bf lp16(const uint16_t* p0, const uint16_t* p1) {
    U128x2 u;
    u.a = *(const uint4*)p0;   // 8 bf16
    u.b = *(const uint4*)p1;   // 8 bf16
    return __builtin_bit_cast(v16bf, u);
}

__device__ __forceinline__ v8f wmma_bf16(v16bf a, v16bf b, v8f c) {
    return __builtin_amdgcn_wmma_f32_16x16x32_bf16(
        /*neg_a=*/false, a, /*neg_b=*/false, b,
        /*c_mod=*/(short)0, c, /*reuse_a=*/false, /*reuse_b=*/false);
}

// ---------- 1) weight fp32 -> bf16 transposed (WT[n*H+k] = W[k*H+n]) ----------

__global__ void wtrans_kernel(const float* __restrict__ W, uint16_t* __restrict__ WT) {
    size_t idx = (size_t)blockIdx.x * blockDim.x + threadIdx.x;  // n*H + k
    int n = (int)(idx >> 10);
    int k = (int)(idx & (HH - 1));
    WT[idx] = f2bf(W[(size_t)k * HH + n]);
}

// ---------- 2) LayerNorm(x)->bf16 and x->bf16 ----------

__global__ void ln_q_kernel(const float* __restrict__ x,
                            const float* __restrict__ w,
                            const float* __restrict__ bsh,
                            uint16_t* __restrict__ xq,
                            uint16_t* __restrict__ xb) {
    __shared__ float s1[256];
    __shared__ float s2[256];
    int row = blockIdx.x;
    int t = threadIdx.x;
    const float* xr = x + (size_t)row * HH;
    float v[4];
    float s = 0.f, ss = 0.f;
#pragma unroll
    for (int i = 0; i < 4; i++) {
        v[i] = xr[t + i * 256];
        s += v[i];
        ss += v[i] * v[i];
    }
    s1[t] = s; s2[t] = ss;
    __syncthreads();
    for (int st = 128; st > 0; st >>= 1) {
        if (t < st) { s1[t] += s1[t + st]; s2[t] += s2[t + st]; }
        __syncthreads();
    }
    float mean = s1[0] * (1.0f / HH);
    float var  = s2[0] * (1.0f / HH) - mean * mean;
    float rstd = rsqrtf(var + 1e-8f);
#pragma unroll
    for (int i = 0; i < 4; i++) {
        int c = t + i * 256;
        float xn = (v[i] - mean) * rstd;
        xq[(size_t)row * HH + c] = f2bf(xn * w[c] + bsh[c]);
        xb[(size_t)row * HH + c] = f2bf(v[i]);
    }
}

// ---------- 3) bf16 GEMM: C[M=8192,N=1024] = A @ BT^T + bias ----------
// grid = (M/32)*2 blocks, 256 threads (8 waves); wave -> 32x64 strip.
// The 32x1024 A panel (64 KB, shared by all 8 waves) is staged into LDS by a
// single Tensor Data Mover DMA (tensor_load_to_lds, TENSORcnt) when available;
// waves then read A tiles via ds_load_b128, and the global VMEM stream carries
// only B. B is software-pipelined through a register double buffer.

__global__ void __launch_bounds__(256, 2)
gemm_bf16_kernel(const uint16_t* __restrict__ A,
                 const uint16_t* __restrict__ BT,
                 const float* __restrict__ bias,
                 float* __restrict__ outF,
                 uint16_t* __restrict__ outB) {
#if HAVE_TDM
    __shared__ __align__(16) uint16_t sA[32 * HH];   // 64 KB A panel
#endif
    const int lane = threadIdx.x & 31;
    const int wave = threadIdx.x >> 5;
    const int m0 = (blockIdx.x >> 1) * 32;
    const int n0 = (blockIdx.x & 1) * 512 + wave * 64;
    const int lr = lane & 15;
    const int ko = (lane & 16) ? 8 : 0;   // K-chunk offset per ISA 16-bit A/B layout
    const int rb = (lane & 16) ? 8 : 0;   // C-layout row base

#if HAVE_TDM
    if (threadIdx.x == 0) {
        // Pack D# per CDNA5 ISA 8.3/8.4: count=1, type=2 (image), data_size=1 (2B),
        // 2-D tensor dim0=1024 stride0=1024, tile 1024x32, dest = sA.
        uint64_t ga = (uint64_t)(uintptr_t)(A + (size_t)m0 * HH);
        uint32_t ldsoff = (uint32_t)(uintptr_t)&sA[0];
        v4u g0;
        g0.x = 1u;                                            // count=1, user mode
        g0.y = ldsoff;                                        // lds_addr
        g0.z = (uint32_t)ga;                                  // global_addr[31:0]
        g0.w = ((uint32_t)(ga >> 32) & 0x1FFFFFFu) | (2u << 30); // ga[56:32] | type=2
        v8i g1;
        g1[0] = 0x00010000;          // wg_mask=0, data_size=1 (2 bytes)
        g1[1] = (int)(1024u << 16);  // tensor_dim0[15:0]=1024 (bits 79:64... low half)
        g1[2] = (int)(32u << 16);    // tensor_dim0 hi=0, tensor_dim1[15:0]=32
        g1[3] = (int)(1024u << 16);  // tensor_dim1 hi=0, tile_dim0=1024
        g1[4] = 32;                  // tile_dim1=32, tile_dim2=0
        g1[5] = 1024;                // tensor_dim0_stride[31:0]=1024
        g1[6] = 0;                   // stride hi / dim1_stride lo
        g1[7] = 0;                   // dim1_stride hi
        v4i z = {0, 0, 0, 0};
#if __clang_major__ >= 23
        v8i z8 = {0, 0, 0, 0, 0, 0, 0, 0};
        __builtin_amdgcn_tensor_load_to_lds(g0, g1, z, z, z8, 0);
#else
        __builtin_amdgcn_tensor_load_to_lds(g0, g1, z, z, 0);
#endif
        __builtin_amdgcn_s_wait_tensorcnt(0);
    }
    __syncthreads();
    const uint16_t* ar0 = &sA[0] + (size_t)lr * HH + ko;
    const uint16_t* ar1 = ar0 + (size_t)16 * HH;
#else
    const uint16_t* ar0 = A + (size_t)(m0 + lr) * HH + ko;
    const uint16_t* ar1 = ar0 + (size_t)16 * HH;
#endif
    const uint16_t* br0 = BT + (size_t)(n0 + lr) * HH + ko;
    const uint16_t* br1 = br0 + (size_t)16 * HH;
    const uint16_t* br2 = br0 + (size_t)32 * HH;
    const uint16_t* br3 = br0 + (size_t)48 * HH;

    v8f c00 = {}, c01 = {}, c02 = {}, c03 = {};
    v8f c10 = {}, c11 = {}, c12 = {}, c13 = {};

    // prologue: load K-step 0
    v16bf a0 = lp16(ar0, ar0 + 16);
    v16bf a1 = lp16(ar1, ar1 + 16);
    v16bf b0 = lp16(br0, br0 + 16);
    v16bf b1 = lp16(br1, br1 + 16);
    v16bf b2 = lp16(br2, br2 + 16);
    v16bf b3 = lp16(br3, br3 + 16);

    for (int kk = 0; kk < HH; kk += 32) {
        const int kn = (kk + 32) & (HH - 1);   // branch-free wrap prefetch
        v16bf na0 = lp16(ar0 + kn, ar0 + kn + 16);
        v16bf na1 = lp16(ar1 + kn, ar1 + kn + 16);
        v16bf nb0 = lp16(br0 + kn, br0 + kn + 16);
        v16bf nb1 = lp16(br1 + kn, br1 + kn + 16);
        v16bf nb2 = lp16(br2 + kn, br2 + kn + 16);
        v16bf nb3 = lp16(br3 + kn, br3 + kn + 16);

        c00 = wmma_bf16(a0, b0, c00);
        c01 = wmma_bf16(a0, b1, c01);
        c02 = wmma_bf16(a0, b2, c02);
        c03 = wmma_bf16(a0, b3, c03);
        c10 = wmma_bf16(a1, b0, c10);
        c11 = wmma_bf16(a1, b1, c11);
        c12 = wmma_bf16(a1, b2, c12);
        c13 = wmma_bf16(a1, b3, c13);

        a0 = na0; a1 = na1;
        b0 = nb0; b1 = nb1; b2 = nb2; b3 = nb3;
    }

#pragma unroll
    for (int mi = 0; mi < 2; mi++) {
#pragma unroll
        for (int t = 0; t < 4; t++) {
            v8f c;
            if (mi == 0) c = (t == 0) ? c00 : (t == 1) ? c01 : (t == 2) ? c02 : c03;
            else         c = (t == 0) ? c10 : (t == 1) ? c11 : (t == 2) ? c12 : c13;
            int n = n0 + t * 16 + lr;
            float bv = bias[n];
#pragma unroll
            for (int r = 0; r < 8; r++) {
                float val = c[r] + bv;
                size_t idx = (size_t)(m0 + mi * 16 + rb + r) * HH + n;
                if (outF) outF[idx] = val;
                else      outB[idx] = f2bf(val);
            }
        }
    }
}

// ---------- 4) v bf16 [B*S,H] -> vT bf16 [B,NH,HS,S] ----------

__global__ void vtrans_kernel(const uint16_t* __restrict__ v, uint16_t* __restrict__ vT) {
    size_t idx = (size_t)blockIdx.x * blockDim.x + threadIdx.x; // ((b*NH+h)*HS+d)*S + s
    int s = (int)(idx & (SS - 1));
    size_t rem = idx >> 11;
    int d = (int)(rem & (HSS - 1));
    size_t bh = rem >> 6;
    int h = (int)(bh & (NHH - 1));
    int b = (int)(bh >> 4);
    vT[idx] = v[(size_t)(b * SS + s) * HH + h * HSS + d];
}

// ---------- 5) flash attention: 1 wave per 16-query tile per (b,h) ----------

__global__ void __launch_bounds__(32, 2)
attn_kernel(const uint16_t* __restrict__ q,
            const uint16_t* __restrict__ k,
            const uint16_t* __restrict__ vT,
            const float* __restrict__ mask,
            uint16_t* __restrict__ out) {
    __shared__ __align__(16) float    sS[16 * 32];
    __shared__ __align__(16) uint16_t sP[16 * 32];
    __shared__ float sCorr[16];
    __shared__ float sL[16];

    const int lane = threadIdx.x;            // 0..31
    const int tile = blockIdx.x & 127;       // S/16 tiles
    const int bh   = blockIdx.x >> 7;
    const int h    = bh & (NHH - 1);
    const int b    = bh >> 4;
    const int s0   = tile * 16;
    const int lr   = lane & 15;
    const int ko   = (lane & 16) ? 8 : 0;
    const int rb   = (lane & 16) ? 8 : 0;

    // Q A-tiles (16x64 = two 16x32), kept in registers
    const uint16_t* qrow = q + (size_t)(b * SS + s0 + lr) * HH + h * HSS + ko;
    v16bf qa0 = lp16(qrow,      qrow + 16);
    v16bf qa1 = lp16(qrow + 32, qrow + 48);

    v8f o0 = {}, o1 = {}, o2 = {}, o3 = {};
    float mrow = -1e30f, lrow = 0.f;         // valid on lanes 0..15 (row = lane)

    const float*    maskB = mask + (size_t)b * SS * SS;
    const uint16_t* kbp   = k  + (size_t)(b * SS) * HH + h * HSS + ko;
    const uint16_t* vbp   = vT + (size_t)(b * NHH + h) * HSS * SS;

    // prologue: load K/V tiles for kb = 0
    const uint16_t* kr0 = kbp + (size_t)lr * HH;
    const uint16_t* kr1 = kbp + (size_t)(16 + lr) * HH;
    v16bf kc00 = lp16(kr0,      kr0 + 16);
    v16bf kc01 = lp16(kr0 + 32, kr0 + 48);
    v16bf kc10 = lp16(kr1,      kr1 + 16);
    v16bf kc11 = lp16(kr1 + 32, kr1 + 48);
    const uint16_t* vr = vbp + (size_t)lr * SS + ko;
    v16bf vc0 = lp16(vr,               vr + 16);
    v16bf vc1 = lp16(vr + 16 * SS,     vr + 16 * SS + 16);
    v16bf vc2 = lp16(vr + 32 * SS,     vr + 32 * SS + 16);
    v16bf vc3 = lp16(vr + 48 * SS,     vr + 48 * SS + 16);

    for (int kb = 0; kb < SS; kb += 32) {
        const int kn = (kb + 32) & (SS - 1);   // branch-free wrap prefetch

        // prefetch next step's K and V tiles
        const uint16_t* nkr0 = kbp + (size_t)(kn + lr) * HH;
        const uint16_t* nkr1 = kbp + (size_t)(kn + 16 + lr) * HH;
        v16bf nk00 = lp16(nkr0,      nkr0 + 16);
        v16bf nk01 = lp16(nkr0 + 32, nkr0 + 48);
        v16bf nk10 = lp16(nkr1,      nkr1 + 16);
        v16bf nk11 = lp16(nkr1 + 32, nkr1 + 48);
        const uint16_t* nvr = vbp + (size_t)lr * SS + kn + ko;
        v16bf nv0 = lp16(nvr,               nvr + 16);
        v16bf nv1 = lp16(nvr + 16 * SS,     nvr + 16 * SS + 16);
        v16bf nv2 = lp16(nvr + 32 * SS,     nvr + 32 * SS + 16);
        v16bf nv3 = lp16(nvr + 48 * SS,     nvr + 48 * SS + 16);

        // scores = Q K^T for 32 keys (two 16x16 C tiles, K=64)
        v8f sc0 = {}, sc1 = {};
        sc0 = wmma_bf16(qa0, kc00, sc0);
        sc0 = wmma_bf16(qa1, kc01, sc0);
        sc1 = wmma_bf16(qa0, kc10, sc1);
        sc1 = wmma_bf16(qa1, kc11, sc1);

        // scale + mask, spill C-layout scores to LDS
#pragma unroll
        for (int r = 0; r < 8; r++) {
            int m = rb + r;
            const float* mp = maskB + (size_t)(s0 + m) * SS + kb;
            sS[m * 32 + lr]      = sc0[r] * 0.125f + mp[lr];
            sS[m * 32 + 16 + lr] = sc1[r] * 0.125f + mp[16 + lr];
        }
        __syncthreads();

        // online softmax: lanes 0..15 each own one row; two float4 passes
        if (lane < 16) {
            const float4* rowp = (const float4*)(sS + lane * 32);
            float tmax = -1e30f;
#pragma unroll
            for (int j = 0; j < 8; j++) {
                float4 vv = rowp[j];
                tmax = fmaxf(tmax, fmaxf(fmaxf(vv.x, vv.y), fmaxf(vv.z, vv.w)));
            }
            float mnew = fmaxf(mrow, tmax);
            float corr = __expf(mrow - mnew);
            float sum = 0.f;
            uint2* prow = (uint2*)(sP + lane * 32);
#pragma unroll
            for (int j = 0; j < 8; j++) {
                float4 vv = rowp[j];
                float p0 = __expf(vv.x - mnew);
                float p1 = __expf(vv.y - mnew);
                float p2 = __expf(vv.z - mnew);
                float p3 = __expf(vv.w - mnew);
                sum += (p0 + p1) + (p2 + p3);
                uint2 pk;
                pk.x = (uint32_t)f2bf(p0) | ((uint32_t)f2bf(p1) << 16);
                pk.y = (uint32_t)f2bf(p2) | ((uint32_t)f2bf(p3) << 16);
                prow[j] = pk;
            }
            lrow = lrow * corr + sum;
            mrow = mnew;
            sCorr[lane] = corr;
        }
        __syncthreads();

        // rescale accumulators by per-row correction
#pragma unroll
        for (int r = 0; r < 8; r++) {
            float cf = sCorr[rb + r];
            o0[r] *= cf; o1[r] *= cf; o2[r] *= cf; o3[r] *= cf;
        }

        // probs A-tile from LDS (row-major 16x32 bf16 -> A layout)
        const uint16_t* pr = sP + lr * 32 + ko;
        v16bf pa = lp16(pr, pr + 16);

        // out += P V (K=32 keys, N=64 head dim)
        o0 = wmma_bf16(pa, vc0, o0);
        o1 = wmma_bf16(pa, vc1, o1);
        o2 = wmma_bf16(pa, vc2, o2);
        o3 = wmma_bf16(pa, vc3, o3);
        __syncthreads();   // sS/sP reused next iteration

        kc00 = nk00; kc01 = nk01; kc10 = nk10; kc11 = nk11;
        vc0 = nv0; vc1 = nv1; vc2 = nv2; vc3 = nv3;
    }

    if (lane < 16) sL[lane] = lrow;
    __syncthreads();
#pragma unroll
    for (int r = 0; r < 8; r++) {
        float inv = 1.f / sL[rb + r];
        size_t rowoff = (size_t)(b * SS + s0 + rb + r) * HH + h * HSS;
        out[rowoff + 0 * 16 + lr]  = f2bf(o0[r] * inv);
        out[rowoff + 1 * 16 + lr]  = f2bf(o1[r] * inv);
        out[rowoff + 2 * 16 + lr]  = f2bf(o2[r] * inv);
        out[rowoff + 3 * 16 + lr]  = f2bf(o3[r] * inv);
    }
}

// ---------- 6) residual + final LayerNorm -> f32 out ----------

__global__ void ln_f_kernel(const float* __restrict__ proj,
                            const float* __restrict__ x,
                            const float* __restrict__ w,
                            const float* __restrict__ bsh,
                            float* __restrict__ out) {
    __shared__ float s1[256];
    __shared__ float s2[256];
    int row = blockIdx.x;
    int t = threadIdx.x;
    const float* pr = proj + (size_t)row * HH;
    const float* xr = x + (size_t)row * HH;
    float v[4];
    float s = 0.f, ss = 0.f;
#pragma unroll
    for (int i = 0; i < 4; i++) {
        int c = t + i * 256;
        v[i] = pr[c] + xr[c];
        s += v[i];
        ss += v[i] * v[i];
    }
    s1[t] = s; s2[t] = ss;
    __syncthreads();
    for (int st = 128; st > 0; st >>= 1) {
        if (t < st) { s1[t] += s1[t + st]; s2[t] += s2[t + st]; }
        __syncthreads();
    }
    float mean = s1[0] * (1.0f / HH);
    float var  = s2[0] * (1.0f / HH) - mean * mean;
    float rstd = rsqrtf(var + 1e-8f);
#pragma unroll
    for (int i = 0; i < 4; i++) {
        int c = t + i * 256;
        out[(size_t)row * HH + c] = w[c] * ((v[i] - mean) * rstd) + bsh[c];
    }
}

// ---------- launch ----------

extern "C" void kernel_launch(void* const* d_in, const int* in_sizes, int n_in,
                              void* d_out, int out_size, void* d_ws, size_t ws_size,
                              hipStream_t stream) {
    (void)in_sizes; (void)n_in; (void)out_size; (void)ws_size;

    const float* x    = (const float*)d_in[0];
    const float* mask = (const float*)d_in[1];
    const float* Wq   = (const float*)d_in[2];
    const float* bq   = (const float*)d_in[3];
    const float* Wk   = (const float*)d_in[4];
    const float* bk   = (const float*)d_in[5];
    const float* Wv   = (const float*)d_in[6];
    const float* bv   = (const float*)d_in[7];
    const float* Wo   = (const float*)d_in[8];
    const float* bo   = (const float*)d_in[9];
    const float* lnqw = (const float*)d_in[10];
    const float* lnqb = (const float*)d_in[11];
    const float* lnfw = (const float*)d_in[12];
    const float* lnfb = (const float*)d_in[13];
    float* outp = (float*)d_out;

    char* ws = (char*)d_ws;
    size_t off = 0;
    auto alloc = [&](size_t bytes) -> void* {
        void* p = ws + off;
        off = (off + bytes + 255) & ~(size_t)255;
        return p;
    };

    uint16_t* wqT  = (uint16_t*)alloc((size_t)HH * HH * 2);
    uint16_t* wkT  = (uint16_t*)alloc((size_t)HH * HH * 2);
    uint16_t* wvT  = (uint16_t*)alloc((size_t)HH * HH * 2);
    uint16_t* woT  = (uint16_t*)alloc((size_t)HH * HH * 2);
    uint16_t* xqb  = (uint16_t*)alloc((size_t)MM * HH * 2);
    uint16_t* xbb  = (uint16_t*)alloc((size_t)MM * HH * 2);
    uint16_t* qb   = (uint16_t*)alloc((size_t)MM * HH * 2);
    uint16_t* kbuf = (uint16_t*)alloc((size_t)MM * HH * 2);
    uint16_t* vbuf = (uint16_t*)alloc((size_t)MM * HH * 2);
    uint16_t* vTb  = (uint16_t*)alloc((size_t)MM * HH * 2);
    uint16_t* attn = (uint16_t*)alloc((size_t)MM * HH * 2);
    float*    proj = (float*)   alloc((size_t)MM * HH * 4);

    const int wtBlocks = (HH * HH) / 256;
    wtrans_kernel<<<wtBlocks, 256, 0, stream>>>(Wq, wqT);
    wtrans_kernel<<<wtBlocks, 256, 0, stream>>>(Wk, wkT);
    wtrans_kernel<<<wtBlocks, 256, 0, stream>>>(Wv, wvT);
    wtrans_kernel<<<wtBlocks, 256, 0, stream>>>(Wo, woT);

    ln_q_kernel<<<MM, 256, 0, stream>>>(x, lnqw, lnqb, xqb, xbb);

    const int gemmBlocks = (MM / 32) * 2;
    gemm_bf16_kernel<<<gemmBlocks, 256, 0, stream>>>(xqb, wqT, bq, nullptr, qb);
    gemm_bf16_kernel<<<gemmBlocks, 256, 0, stream>>>(xbb, wkT, bk, nullptr, kbuf);
    gemm_bf16_kernel<<<gemmBlocks, 256, 0, stream>>>(xbb, wvT, bv, nullptr, vbuf);

    vtrans_kernel<<<(MM * HH) / 256, 256, 0, stream>>>(vbuf, vTb);

    attn_kernel<<<BB * NHH * (SS / 16), 32, 0, stream>>>(qb, kbuf, vTb, mask, attn);

    gemm_bf16_kernel<<<gemmBlocks, 256, 0, stream>>>(attn, woT, bo, proj, nullptr);

    ln_f_kernel<<<MM, 256, 0, stream>>>(proj, x, lnfw, lnfb, outp);
}
